// WindowAttention_20590073217242
// MI455X (gfx1250) — matrix operand order repaired and checked by reading, not compiled, not run on hardware
//
#include <hip/hip_runtime.h>
#include <hip/hip_bf16.h>

// ---------------------------------------------------------------------------
// Swin window attention, MI455X (gfx1250, wave32, WMMA f32_16x16x32_f16)
//   DIM=512, HEADS=16, HD=32, S=49 (padded to 64), B=2048 windows, NW=4
// Pipeline:
//   0a) x f32 -> f16            (makes A panel L2-resident: 103 MB < 192 MB L2)
//   0b) w_qkv, w_proj -> f16 transposed [n][k]   (B fragments = contiguous)
//   1)  QKV GEMM  (B-panel staged once in LDS via async global->LDS b128)
//   2)  per-(window,head) attention, register softmax
//   3)  proj GEMM
// ---------------------------------------------------------------------------

#define USE_ASYNC_LDS 1

typedef _Float16 h8  __attribute__((ext_vector_type(8)));
typedef _Float16 h16 __attribute__((ext_vector_type(16)));
typedef float    f8  __attribute__((ext_vector_type(8)));

__device__ __forceinline__ h16 h16_zero() {
    h16 z;
#pragma unroll
    for (int i = 0; i < 16; ++i) z[i] = (_Float16)0;
    return z;
}
__device__ __forceinline__ f8 f8_zero() {
    f8 z;
#pragma unroll
    for (int i = 0; i < 8; ++i) z[i] = 0.0f;
    return z;
}
__device__ __forceinline__ h16 h16_cat(h8 lo, h8 hi) {
    return __builtin_shufflevector(lo, hi, 0,1,2,3,4,5,6,7,8,9,10,11,12,13,14,15);
}

// D = A(16x32 f16) * B(32x16 f16) + C(16x16 f32)
__device__ __forceinline__ f8 wmma_f16(h16 a, h16 b, f8 c) {
    return __builtin_amdgcn_wmma_f32_16x16x32_f16(false, a, false, b, (short)0, c,
                                                  false, false);
}

// --- async global -> LDS copy (CDNA5), 16 bytes per lane --------------------
__device__ __forceinline__ void async_copy_b128(void* lds, const void* g) {
#if USE_ASYNC_LDS
    // ISA 10.2: flat shared addr low 32 bits == LDS byte offset
    unsigned int l = (unsigned int)(unsigned long long)lds;
    asm volatile("global_load_async_to_lds_b128 %0, %1, off"
                 :: "v"(l), "v"(g) : "memory");
#else
    *(h8*)lds = *(const h8*)g;
#endif
}
__device__ __forceinline__ void async_wait_all() {
#if USE_ASYNC_LDS
    asm volatile("s_wait_asynccnt 0x0" ::: "memory");
#endif
}

// --- A fragment (16x32, f16 row-major memory, row stride ld halfs) ----------
// ISA 7.12.2: lanes 0-15: M=lane, K={0..7,16..23}; lanes 16-31: M=lane-16,
// K={8..15,24..31}.
__device__ __forceinline__ h16 load_a_f16(const _Float16* base, int ld, int lane) {
    const int m    = lane & 15;
    const int koff = (lane < 16) ? 0 : 8;
    const _Float16* p = base + m * ld + koff;
    h8 lo = *(const h8*)(p);
    h8 hi = *(const h8*)(p + 16);
    return h16_cat(lo, hi);
}
__device__ __forceinline__ h16 load_a_f16_guard(const _Float16* base, int ld, int lane,
                                                int row0, int rows) {
    const int m = lane & 15;
    if (row0 + m >= rows) return h16_zero();
    const int koff = (lane < 16) ? 0 : 8;
    const _Float16* p = base + (row0 + m) * ld + koff;
    h8 lo = *(const h8*)(p);
    h8 hi = *(const h8*)(p + 16);
    return h16_cat(lo, hi);
}

// --- B fragment (32x16) from memory holding B transposed: bt[n][k] ----------
// lanes 0-15: n=lane, K={0..7,16..23}; lanes 16-31: n=lane-16, K={8..15,24..31}.
__device__ __forceinline__ h16 load_bt_f16(const _Float16* base, int ld, int lane) {
    const int n    = lane & 15;
    const int koff = (lane < 16) ? 0 : 8;
    const _Float16* p = base + n * ld + koff;
    h8 lo = *(const h8*)(p);
    h8 hi = *(const h8*)(p + 16);
    return h16_cat(lo, hi);
}
__device__ __forceinline__ h16 load_bt_f16_guard(const _Float16* base, int ld, int lane,
                                                 int row0, int rows) {
    const int n = lane & 15;
    if (row0 + n >= rows) return h16_zero();
    const int koff = (lane < 16) ? 0 : 8;
    const _Float16* p = base + (row0 + n) * ld + koff;
    h8 lo = *(const h8*)(p);
    h8 hi = *(const h8*)(p + 16);
    return h16_cat(lo, hi);
}

// ---------------------------------------------------------------------------
// Kernel 0a: f32 -> f16 bulk convert (8 elems/thread)
// ---------------------------------------------------------------------------
__global__ __launch_bounds__(256) void cvt_f16_kernel(
    const float* __restrict__ src, _Float16* __restrict__ dst, int n8) {
    const int i = blockIdx.x * 256 + threadIdx.x;
    if (i >= n8) return;
    const float4* s = (const float4*)src + (size_t)i * 2;
    float4 f0 = s[0], f1 = s[1];
    h8 v;
    v[0]=(_Float16)f0.x; v[1]=(_Float16)f0.y; v[2]=(_Float16)f0.z; v[3]=(_Float16)f0.w;
    v[4]=(_Float16)f1.x; v[5]=(_Float16)f1.y; v[6]=(_Float16)f1.z; v[7]=(_Float16)f1.w;
    *((h8*)dst + i) = v;
}

// ---------------------------------------------------------------------------
// Kernel 0b: dst[n*K + k] = (f16)src[k*N + n]; 64x64 LDS tile transpose
// grid (K/64, N/64) x 256 threads
// ---------------------------------------------------------------------------
__global__ __launch_bounds__(256) void transpose_cvt_kernel(
    const float* __restrict__ src, _Float16* __restrict__ dst, int K, int N) {
    __shared__ _Float16 tile[64 * 72];
    const int k0 = blockIdx.x * 64, n0 = blockIdx.y * 64;
    const int tx = threadIdx.x & 63, ty = threadIdx.x >> 6;  // ty in 0..3
#pragma unroll
    for (int i = 0; i < 16; ++i) {
        const int kk = ty + i * 4;
        tile[kk * 72 + tx] = (_Float16)src[(size_t)(k0 + kk) * N + n0 + tx];
    }
    __syncthreads();
#pragma unroll
    for (int i = 0; i < 16; ++i) {
        const int nn = ty + i * 4;
        dst[(size_t)(n0 + nn) * K + k0 + tx] = tile[tx * 72 + nn];
    }
}

// ---------------------------------------------------------------------------
// Kernel 1: qkv = (xh @ w_qkv + b_qkv), q pre-scaled, f16 out [B*S, 1536]
// grid (784, 24) x 256 threads; WG tile 128(M) x 64(N); B panel in LDS once.
// ---------------------------------------------------------------------------
__global__ __launch_bounds__(256) void qkv_gemm_kernel(
    const _Float16* __restrict__ xh, const _Float16* __restrict__ wT,
    const float* __restrict__ bvec, _Float16* __restrict__ out) {
    __shared__ _Float16 wt[64 * 520];  // [n][k], ld=520 halfs (16B-aligned rows)
    const int tid   = threadIdx.x;
    const int lane  = tid & 31;
    const int wid   = tid >> 5;
    const int mbase = blockIdx.x * 128;
    const int nbase = blockIdx.y * 64;

    // stage B panel: 64 rows x 1024B, async b128 (16B/lane x 16 iters)
#pragma unroll
    for (int it = 0; it < 16; ++it) {
        const int e   = it * 256 + tid;  // 16B units
        const int n   = e >> 6;
        const int seg = e & 63;
        async_copy_b128(&wt[n * 520 + seg * 8],
                        wT + (size_t)(nbase + n) * 512 + seg * 8);
    }
    async_wait_all();
    __syncthreads();

    f8 acc[4];
#pragma unroll
    for (int t = 0; t < 4; ++t) acc[t] = f8_zero();

    const _Float16* arow = xh + (size_t)(mbase + wid * 16) * 512;
    for (int k0 = 0; k0 < 512; k0 += 32) {
        __builtin_prefetch(arow + k0 + 64, 0, 1);
        h16 a = load_a_f16(arow + k0, 512, lane);
#pragma unroll
        for (int t = 0; t < 4; ++t) {
            h16 b = load_bt_f16(wt + t * 16 * 520 + k0, 520, lane);
            acc[t] = wmma_f16(a, b, acc[t]);
        }
    }

    const float qscale = 0.17677669529663687f;  // 1/sqrt(32)
    const int rowadd = (lane < 16) ? 0 : 8;
    const int ncol   = lane & 15;
#pragma unroll
    for (int t = 0; t < 4; ++t) {
        const int col  = nbase + t * 16 + ncol;
        const float bb = bvec[col];
        const float sc = (col < 512) ? qscale : 1.0f;
#pragma unroll
        for (int r = 0; r < 8; ++r) {
            const int m = mbase + wid * 16 + r + rowadd;
            out[(size_t)m * 1536 + col] = (_Float16)((acc[t][r] + bb) * sc);
        }
    }
}

// ---------------------------------------------------------------------------
// Kernel 2: per (window b, head h): P = softmax(QK^T + bias + mask); O = P@V
// grid 2048*16 blocks x 128 threads (4 waves); wave w owns query rows 16w..
// ---------------------------------------------------------------------------
__global__ __launch_bounds__(128) void attn_kernel(
    const _Float16* __restrict__ qkv, const float* __restrict__ mask,
    const float* __restrict__ bias_table, const int* __restrict__ rel_idx,
    _Float16* __restrict__ o) {
    __shared__ _Float16 vt[32 * 72];        // V^T: [d][key], ld=72
    __shared__ _Float16 pbuf[4 * 16 * 72];  // per-wave softmax(P) f16, ld=72

    const int tid  = threadIdx.x;
    const int lane = tid & 31;
    const int wid  = tid >> 5;
    const int b    = blockIdx.x >> 4;
    const int h    = blockIdx.x & 15;

    const _Float16* qp = qkv + (size_t)(b * 49) * 1536 + h * 32;
    const _Float16* kp = qp + 512;
    const _Float16* vp = qp + 1024;

    // stage V transposed into LDS (zero-pad keys 49..63)
#pragma unroll
    for (int e = tid; e < 32 * 64; e += 128) {
        const int d   = e & 31;
        const int key = e >> 5;
        vt[d * 72 + key] = (key < 49) ? vp[(size_t)key * 1536 + d] : (_Float16)0;
    }

    // ---- logits P = Q @ K^T (K=hd=32: one WMMA per 16x16 tile)
    const int mrow = wid * 16;
    h16 aq = load_a_f16_guard(qp, 1536, lane, mrow, 49);
    f8 p[4];
#pragma unroll
    for (int t = 0; t < 4; ++t) {
        p[t] = f8_zero();
        h16 bk = load_bt_f16_guard(kp, 1536, lane, t * 16, 49);
        p[t] = wmma_f16(aq, bk, p[t]);
    }

    // ---- + relative-position bias + mask; pad -> -inf (C-layout addressing)
    const float* mk = mask + (size_t)(b & 3) * (49 * 49);
    const int rowadd = (lane < 16) ? 0 : 8;
    const int ncol   = lane & 15;
#pragma unroll
    for (int t = 0; t < 4; ++t) {
        const int kcol = t * 16 + ncol;
#pragma unroll
        for (int r = 0; r < 8; ++r) {
            const int qrow = mrow + r + rowadd;
            if (kcol < 49 && qrow < 49) {
                const int ri = rel_idx[qrow * 49 + kcol];
                p[t][r] += bias_table[ri * 16 + h] + mk[qrow * 49 + kcol];
            } else {
                p[t][r] = -3.0e38f;
            }
        }
    }

    // ---- softmax per row, in registers (xor-shuffles stay in 16-lane half)
    _Float16* pw = pbuf + wid * 16 * 72;
#pragma unroll
    for (int r = 0; r < 8; ++r) {
        float mx = fmaxf(fmaxf(p[0][r], p[1][r]), fmaxf(p[2][r], p[3][r]));
#pragma unroll
        for (int off = 1; off < 16; off <<= 1) mx = fmaxf(mx, __shfl_xor(mx, off));
        float ex[4];
#pragma unroll
        for (int t = 0; t < 4; ++t) ex[t] = __expf(p[t][r] - mx);
        float sum = ex[0] + ex[1] + ex[2] + ex[3];
#pragma unroll
        for (int off = 1; off < 16; off <<= 1) sum += __shfl_xor(sum, off);
        const float inv = 1.0f / sum;
        const int rr = r + rowadd;
#pragma unroll
        for (int t = 0; t < 4; ++t)
            pw[rr * 72 + t * 16 + ncol] = (_Float16)(ex[t] * inv);
    }
    __syncthreads();  // publish vt (cross-wave) and pbuf (cross-lane)

    // ---- O = P @ V : A from pbuf (16x64 -> 2 k-steps), B from vt
    h16 a0 = load_a_f16(pw, 72, lane);
    h16 a1 = load_a_f16(pw + 32, 72, lane);
    f8 o0 = f8_zero(), o1 = f8_zero();
    h16 b00 = load_bt_f16(vt, 72, lane);
    h16 b01 = load_bt_f16(vt + 32, 72, lane);
    h16 b10 = load_bt_f16(vt + 16 * 72, 72, lane);
    h16 b11 = load_bt_f16(vt + 16 * 72 + 32, 72, lane);
    o0 = wmma_f16(a0, b00, o0);
    o0 = wmma_f16(a1, b01, o0);
    o1 = wmma_f16(a0, b10, o1);
    o1 = wmma_f16(a1, b11, o1);

#pragma unroll
    for (int r = 0; r < 8; ++r) {
        const int qrow = mrow + r + rowadd;
        if (qrow < 49) {
            const size_t base = (size_t)(b * 49 + qrow) * 512 + h * 32 + ncol;
            o[base]      = (_Float16)o0[r];
            o[base + 16] = (_Float16)o1[r];
        }
    }
}

// ---------------------------------------------------------------------------
// Kernel 3: out = o_h @ w_proj + b_proj  (M=100352, K=512, N=512, f32 out)
// ---------------------------------------------------------------------------
__global__ __launch_bounds__(256) void proj_gemm_kernel(
    const _Float16* __restrict__ a, const _Float16* __restrict__ wT,
    const float* __restrict__ bvec, float* __restrict__ out) {
    __shared__ _Float16 wt[64 * 520];
    const int tid   = threadIdx.x;
    const int lane  = tid & 31;
    const int wid   = tid >> 5;
    const int mbase = blockIdx.x * 128;
    const int nbase = blockIdx.y * 64;

#pragma unroll
    for (int it = 0; it < 16; ++it) {
        const int e   = it * 256 + tid;
        const int n   = e >> 6;
        const int seg = e & 63;
        async_copy_b128(&wt[n * 520 + seg * 8],
                        wT + (size_t)(nbase + n) * 512 + seg * 8);
    }
    async_wait_all();
    __syncthreads();

    f8 acc[4];
#pragma unroll
    for (int t = 0; t < 4; ++t) acc[t] = f8_zero();

    const _Float16* arow = a + (size_t)(mbase + wid * 16) * 512;
    for (int k0 = 0; k0 < 512; k0 += 32) {
        __builtin_prefetch(arow + k0 + 64, 0, 1);
        h16 af = load_a_f16(arow + k0, 512, lane);
#pragma unroll
        for (int t = 0; t < 4; ++t) {
            h16 bf = load_bt_f16(wt + t * 16 * 520 + k0, 520, lane);
            acc[t] = wmma_f16(af, bf, acc[t]);
        }
    }

    const int rowadd = (lane < 16) ? 0 : 8;
    const int ncol   = lane & 15;
#pragma unroll
    for (int t = 0; t < 4; ++t) {
        const int col  = nbase + t * 16 + ncol;
        const float bb = bvec[col];
#pragma unroll
        for (int r = 0; r < 8; ++r) {
            const int m = mbase + wid * 16 + r + rowadd;
            out[(size_t)m * 512 + col] = acc[t][r] + bb;
        }
    }
}

// ---------------------------------------------------------------------------
extern "C" void kernel_launch(void* const* d_in, const int* in_sizes, int n_in,
                              void* d_out, int out_size, void* d_ws, size_t ws_size,
                              hipStream_t stream) {
    const float* x          = (const float*)d_in[0];
    const float* mask       = (const float*)d_in[1];
    const float* w_qkv      = (const float*)d_in[2];
    const float* b_qkv      = (const float*)d_in[3];
    const float* w_proj     = (const float*)d_in[4];
    const float* b_proj     = (const float*)d_in[5];
    const float* bias_table = (const float*)d_in[6];
    const int*   rel_idx    = (const int*)d_in[7];
    float* out = (float*)d_out;

    char* ws = (char*)d_ws;
    const size_t QKV_BYTES = (size_t)2048 * 49 * 1536 * 2;  // 308.3 MB
    const size_t O_BYTES   = (size_t)2048 * 49 * 512 * 2;   // 102.8 MB
    const size_t XH_BYTES  = (size_t)2048 * 49 * 512 * 2;   // 102.8 MB
    const size_t WQT_BYTES = (size_t)1536 * 512 * 2;
    _Float16* qkv_h  = (_Float16*)(ws);
    _Float16* o_h    = (_Float16*)(ws + QKV_BYTES);
    _Float16* xh     = (_Float16*)(ws + QKV_BYTES + O_BYTES);
    _Float16* wqkvT  = (_Float16*)(ws + QKV_BYTES + O_BYTES + XH_BYTES);
    _Float16* wprojT = (_Float16*)(ws + QKV_BYTES + O_BYTES + XH_BYTES + WQT_BYTES);

    // 0a) x -> f16 (51,380,224 elems = 6,422,528 x 8)
    cvt_f16_kernel<<<dim3(25088), 256, 0, stream>>>(x, xh, 6422528);
    // 0b) weights -> f16, transposed [n][k]
    transpose_cvt_kernel<<<dim3(8, 24), 256, 0, stream>>>(w_qkv, wqkvT, 512, 1536);
    transpose_cvt_kernel<<<dim3(8, 8), 256, 0, stream>>>(w_proj, wprojT, 512, 512);
    // 1) QKV projection: [100352,512] x [512,1536]
    qkv_gemm_kernel<<<dim3(784, 24), 256, 0, stream>>>(xh, wqkvT, b_qkv, qkv_h);
    // 2) windowed attention per (window, head)
    attn_kernel<<<dim3(2048 * 16), 128, 0, stream>>>(qkv_h, mask, bias_table,
                                                     rel_idx, o_h);
    // 3) output projection: [100352,512] x [512,512]
    proj_gemm_kernel<<<dim3(784, 8), 256, 0, stream>>>(o_h, wprojT, b_proj, out);
}
